// Recurrence_79448305041479
// MI455X (gfx1250) — matrix operand order repaired and checked by reading, not compiled, not run on hardware
//
#include <hip/hip_runtime.h>
#include <hip/hip_bf16.h>

typedef __attribute__((ext_vector_type(8)))  _Float16 v8h;
typedef __attribute__((ext_vector_type(16))) _Float16 v16h;
typedef __attribute__((ext_vector_type(8)))  float    v8f;

#define TT 64
#define NN 1024
#define LL 128
#define HH 512
#define CC 64
#define SS 771          // 1 + 1 + H + (L+1) + L
#define K0 576          // C + H
#define NB 16           // batch tile per workgroup

// ---------- fragment loaders ----------
// A fragment (16-bit A, 16x32): lane m=lane&15 is row M; half=lane>>4 selects K phase.
// v16h elems 0..15 = K [k0+16*half, +16) (contiguous)
__device__ inline v16h ld_a_frag(const _Float16* arow, int k0, int half) {
  const v8h* lo = (const v8h*)(arow + k0 + 16 * half);
  const v8h* hi = (const v8h*)(arow + k0 + 16 * half + 8);
  v8h a = *lo, b = *hi;
  return __builtin_shufflevector(a, b, 0,1,2,3,4,5,6,7,8,9,10,11,12,13,14,15);
}

// B fragment (16-bit B, 32x16): lane n=lane&15 is column N; 16 contiguous K starting
// at k0 + 16*(lane>>4). Row-major W (out,in) rows are exactly these contiguous halfs.
__device__ inline v16h ld_b_frag(const _Float16* wrow, int off) {
  const v8h* q = (const v8h*)(wrow + off);
  v8h a = q[0], b = q[1];
  return __builtin_shufflevector(a, b, 0,1,2,3,4,5,6,7,8,9,10,11,12,13,14,15);
}

// ---------- fused GEMM + bias + ReLU + f16 store ----------
template <int K, int ASTRIDE, int OSTRIDE>
__device__ inline void gemm_relu(const _Float16* __restrict__ Wg,
                                 const float* __restrict__ bias,
                                 const _Float16* __restrict__ A,   // LDS
                                 _Float16* __restrict__ O,         // LDS
                                 int wid, int mrow, int half) {
  const _Float16* arow = A + mrow * ASTRIDE;
  const int kb = 16 * half;
  for (int q = 0; q < 4; ++q) {
    const int jt   = wid * 4 + q;
    const int ncol = jt * 16 + mrow;
    const _Float16* wrow = Wg + (size_t)ncol * K;
    __builtin_prefetch(wrow, 0, 0);
    v8f acc = {0.f,0.f,0.f,0.f,0.f,0.f,0.f,0.f};
    for (int kc = 0; kc < K / 32; ++kc) {
      v16h a = ld_a_frag(arow, kc * 32, half);
      v16h b = ld_b_frag(wrow, kc * 32 + kb);
      acc = __builtin_amdgcn_wmma_f32_16x16x32_f16(false, a, false, b,
                                                   (short)0, acc, false, false);
    }
    const float bj = bias[ncol];
#pragma unroll
    for (int r = 0; r < 8; ++r) {
      float v = acc[r] + bj;
      v = v > 0.f ? v : 0.f;
      O[(r + 8 * half) * OSTRIDE + ncol] = (_Float16)v;
    }
  }
}

__device__ inline float sigmoidf_(float x) { return 1.f / (1.f + expf(-x)); }

__global__ __launch_bounds__(256)
void recurrence_kernel(const float* __restrict__ cond,
                       const float* __restrict__ rnn,
                       const float* __restrict__ emb,
                       const float* __restrict__ b0,
                       const float* __restrict__ b1,
                       const float* __restrict__ b2,
                       const float* __restrict__ b_ih,
                       const float* __restrict__ b_hh,
                       const float* __restrict__ Wc,
                       const float* __restrict__ bc,
                       const float* __restrict__ Wa,
                       const float* __restrict__ ba,
                       const int*   __restrict__ lines,
                       const int*   __restrict__ actions,
                       const _Float16* __restrict__ W0h,
                       const _Float16* __restrict__ W1h,
                       const _Float16* __restrict__ W2h,
                       const _Float16* __restrict__ Wihh,
                       const _Float16* __restrict__ Whhh,
                       float* __restrict__ out) {
  __shared__ alignas(16) _Float16 xA[NB * K0];   // 18432 B
  __shared__ alignas(16) _Float16 xB[NB * HH];   // 16384 B
  __shared__ alignas(16) _Float16 hF[NB * HH];   // 16384 B
  __shared__ float kv[NB * 5];
  __shared__ float heads[NB * 5];                // l0,l1,l2,no_op,v
  __shared__ int   idxs[NB];

  const int tid  = threadIdx.x;
  const int wid  = tid >> 5;        // wave32
  const int lane = tid & 31;
  const int mrow = lane & 15;
  const int half = lane >> 4;
  const int n0   = blockIdx.x * NB;

  // ---- init: h0 from rnn_hxs[:, 2:2+H]; idx from one-hot p0 (zeros -> idx 0) ----
  for (int e = tid; e < NB * HH; e += 256) {
    int s = e >> 9, j = e & 511;
    hF[e] = (_Float16)rnn[(size_t)(n0 + s) * SS + 2 + j];
  }
  if (tid < NB) {
    int idx = 0;
    const float* pr = rnn + (size_t)(n0 + tid) * SS + 2 + HH + (LL + 1);
    for (int l = 0; l < LL; ++l) { if (pr[l] > 0.5f) { idx = l; break; } }
    idxs[tid] = idx;
  }
  __syncthreads();

  for (int t = 0; t < TT; ++t) {
    // ---- stage x = [cond_t | r], r = emb[lines[n, idx_n]] (p one-hot) ----
    for (int e = tid; e < NB * K0; e += 256) {
      int s = e / K0, c = e % K0;
      float v;
      if (c < CC) {
        v = cond[((size_t)t * NN + n0 + s) * CC + c];
      } else {
        int row = lines[(n0 + s) * LL + idxs[s]];
        v = emb[(size_t)row * HH + (c - CC)];
      }
      xA[e] = (_Float16)v;
    }
    __syncthreads();

    gemm_relu<K0, K0, HH>(W0h, b0, xA, xB, wid, mrow, half);   // x0 -> x1
    __syncthreads();
    gemm_relu<HH, HH, K0>(W1h, b1, xB, xA, wid, mrow, half);   // x1 -> x2
    __syncthreads();
    gemm_relu<HH, K0, HH>(W2h, b2, xA, xB, wid, mrow, half);   // x2 -> x3
    __syncthreads();

    // ---- fused GRU: gi = x3@W_ih^T, gh = h@W_hh^T, elementwise in C-layout ----
    float hn[4][8];
    {
      const _Float16* axrow = xB + mrow * HH;
      const _Float16* ahrow = hF + mrow * HH;
      const int kb = 16 * half;
      for (int q = 0; q < 4; ++q) {
        const int jt   = wid * 4 + q;
        const int ncol = jt * 16 + mrow;
        const _Float16* wir = Wihh + (size_t)(ncol)        * HH;
        const _Float16* wiz = Wihh + (size_t)(HH  + ncol)  * HH;
        const _Float16* win = Wihh + (size_t)(2*HH + ncol) * HH;
        const _Float16* whr = Whhh + (size_t)(ncol)        * HH;
        const _Float16* whz = Whhh + (size_t)(HH  + ncol)  * HH;
        const _Float16* whn = Whhh + (size_t)(2*HH + ncol) * HH;
        __builtin_prefetch(wir, 0, 0);
        __builtin_prefetch(whr, 0, 0);
        v8f ar = {0.f,0.f,0.f,0.f,0.f,0.f,0.f,0.f};
        v8f az = ar, an = ar, hr = ar, hz = ar, hnn = ar;
        for (int kc = 0; kc < HH / 32; ++kc) {
          const int k0 = kc * 32;
          v16h ax = ld_a_frag(axrow, k0, half);
          v16h ah = ld_a_frag(ahrow, k0, half);
          v16h b;
          b = ld_b_frag(wir, k0 + kb);
          ar  = __builtin_amdgcn_wmma_f32_16x16x32_f16(false, ax, false, b, (short)0, ar,  false, false);
          b = ld_b_frag(wiz, k0 + kb);
          az  = __builtin_amdgcn_wmma_f32_16x16x32_f16(false, ax, false, b, (short)0, az,  false, false);
          b = ld_b_frag(win, k0 + kb);
          an  = __builtin_amdgcn_wmma_f32_16x16x32_f16(false, ax, false, b, (short)0, an,  false, false);
          b = ld_b_frag(whr, k0 + kb);
          hr  = __builtin_amdgcn_wmma_f32_16x16x32_f16(false, ah, false, b, (short)0, hr,  false, false);
          b = ld_b_frag(whz, k0 + kb);
          hz  = __builtin_amdgcn_wmma_f32_16x16x32_f16(false, ah, false, b, (short)0, hz,  false, false);
          b = ld_b_frag(whn, k0 + kb);
          hnn = __builtin_amdgcn_wmma_f32_16x16x32_f16(false, ah, false, b, (short)0, hnn, false, false);
        }
        const float bir = b_ih[ncol], biz = b_ih[HH + ncol], bin_ = b_ih[2*HH + ncol];
        const float bhr = b_hh[ncol], bhz = b_hh[HH + ncol], bhn_ = b_hh[2*HH + ncol];
#pragma unroll
        for (int r = 0; r < 8; ++r) {
          float rg = sigmoidf_((ar[r] + bir) + (hr[r] + bhr));
          float z  = sigmoidf_((az[r] + biz) + (hz[r] + bhz));
          float nv = tanhf((an[r] + bin_) + rg * (hnn[r] + bhn_));
          float ho = (float)hF[(r + 8 * half) * HH + ncol];
          hn[q][r] = (1.f - z) * nv + z * ho;
        }
      }
    }
    __syncthreads();   // all reads of old h done
    for (int q = 0; q < 4; ++q) {
      const int ncol = (wid * 4 + q) * 16 + mrow;
#pragma unroll
      for (int r = 0; r < 8; ++r)
        hF[(r + 8 * half) * HH + ncol] = (_Float16)hn[q][r];
    }
    __syncthreads();

    // ---- tiny heads: k = h@Wa^T + ba (4), v = h@Wc^T + bc (1) ----
    if (tid < NB * 5) {
      const int s = tid / 5, o = tid % 5;
      const float* w = (o < 4) ? (Wa + o * HH) : Wc;
      const _Float16* hrow = hF + s * HH;
      float acc = 0.f;
      for (int k = 0; k < HH; ++k) acc += (float)hrow[k] * w[k];
      acc += (o < 4) ? ba[o] : bc[0];
      kv[tid] = acc;
    }
    __syncthreads();
    if (tid < NB) {
      float k0v = kv[tid*5+0], k1v = kv[tid*5+1], k2v = kv[tid*5+2];
      float mx = fmaxf(k0v, fmaxf(k1v, k2v));
      float e0 = expf(k0v - mx), e1 = expf(k1v - mx), e2 = expf(k2v - mx);
      float es = e0 + e1 + e2;
      heads[tid*5+0] = e0 / es;
      heads[tid*5+1] = e1 / es;
      heads[tid*5+2] = e2 / es;
      heads[tid*5+3] = sigmoidf_(kv[tid*5+3]);
      heads[tid*5+4] = kv[tid*5+4];
    }
    __syncthreads();

    // ---- outputs: [a, v, h(512), probs(129), p(128)] (+ dup of last step) ----
    const int reps = (t == TT - 1) ? 2 : 1;
    for (int rep = 0; rep < reps; ++rep) {
      const size_t tb = (rep ? (size_t)TT : (size_t)t) * NN;
      for (int e = tid; e < NB * HH; e += 256) {          // h copy (coalesced in j)
        int s = e >> 9, j = e & 511;
        out[(tb + n0 + s) * SS + 2 + j] = (float)hF[e];
      }
      for (int e = tid; e < NB * 257; e += 256) {         // zero probs+p region
        int s = e / 257, j = e % 257;
        out[(tb + n0 + s) * SS + 514 + j] = 0.f;
      }
    }
    __syncthreads();
    if (tid < NB) {                                       // sparse scatter
      const int s = tid, nn = n0 + s;
      const float l0 = heads[s*5+0], l1 = heads[s*5+1], l2 = heads[s*5+2];
      const float noop = heads[s*5+3], vv = heads[s*5+4];
      const int a = actions[t * NN + nn];
      const int idx = idxs[s];
      const float sum = (l0 + l1 + l2) * (1.f - noop) + noop;
      const float sc  = (1.f - noop) / sum;
      const int nidx = (a < LL) ? a : idx;
      for (int rep = 0; rep < reps; ++rep) {
        const size_t base = ((rep ? (size_t)TT : (size_t)t) * NN + nn) * SS;
        out[base + 0] = (float)a;
        out[base + 1] = vv;
        if (idx == 0)            { out[base+514+0]    = (l1+l2)*sc; out[base+514+1]    = l0*sc; }
        else if (idx == LL - 1)  { out[base+514+LL-2] = l2*sc;      out[base+514+LL-1] = (l0+l1)*sc; }
        else { out[base+514+idx-1] = l2*sc; out[base+514+idx] = l1*sc; out[base+514+idx+1] = l0*sc; }
        out[base + 514 + LL] = noop / sum;
        out[base + 643 + nidx] = 1.0f;
      }
    }
    __syncthreads();
    if (tid < NB) {                                       // advance one-hot index
      const int a = actions[t * NN + n0 + tid];
      if (a < LL) idxs[tid] = a;
    }
    __syncthreads();
  }
}

// ---------- one-time f32 -> f16 weight conversion ----------
__global__ void cvt_f32_to_f16(const float* __restrict__ src,
                               _Float16* __restrict__ dst, int n) {
  int i = blockIdx.x * blockDim.x + threadIdx.x;
  if (i < n) dst[i] = (_Float16)src[i];
}

extern "C" void kernel_launch(void* const* d_in, const int* in_sizes, int n_in,
                              void* d_out, int out_size, void* d_ws, size_t ws_size,
                              hipStream_t stream) {
  const float* cond = (const float*)d_in[0];
  const float* rnn  = (const float*)d_in[1];
  const float* emb  = (const float*)d_in[2];
  const float* W0   = (const float*)d_in[3];
  const float* b0   = (const float*)d_in[4];
  const float* W1   = (const float*)d_in[5];
  const float* b1   = (const float*)d_in[6];
  const float* W2   = (const float*)d_in[7];
  const float* b2   = (const float*)d_in[8];
  const float* Wih  = (const float*)d_in[9];
  const float* bih  = (const float*)d_in[10];
  const float* Whh  = (const float*)d_in[11];
  const float* bhh  = (const float*)d_in[12];
  const float* Wc   = (const float*)d_in[13];
  const float* bc   = (const float*)d_in[14];
  const float* Wa   = (const float*)d_in[15];
  const float* ba   = (const float*)d_in[16];
  const int* lines   = (const int*)d_in[17];
  const int* actions = (const int*)d_in[18];
  float* out = (float*)d_out;

  _Float16* ws = (_Float16*)d_ws;
  const size_t n0 = (size_t)HH * K0;       // W0: 512x576
  const size_t n1 = (size_t)HH * HH;       // W1
  const size_t n2 = (size_t)HH * HH;       // W2
  const size_t n3 = (size_t)3 * HH * HH;   // W_ih
  const size_t n4 = (size_t)3 * HH * HH;   // W_hh
  _Float16* W0h  = ws;
  _Float16* W1h  = W0h + n0;
  _Float16* W2h  = W1h + n1;
  _Float16* Wihh = W2h + n2;
  _Float16* Whhh = Wihh + n3;

  cvt_f32_to_f16<<<(int)((n0 + 255) / 256), 256, 0, stream>>>(W0,  W0h,  (int)n0);
  cvt_f32_to_f16<<<(int)((n1 + 255) / 256), 256, 0, stream>>>(W1,  W1h,  (int)n1);
  cvt_f32_to_f16<<<(int)((n2 + 255) / 256), 256, 0, stream>>>(W2,  W2h,  (int)n2);
  cvt_f32_to_f16<<<(int)((n3 + 255) / 256), 256, 0, stream>>>(Wih, Wihh, (int)n3);
  cvt_f32_to_f16<<<(int)((n4 + 255) / 256), 256, 0, stream>>>(Whh, Whhh, (int)n4);

  recurrence_kernel<<<NN / NB, 256, 0, stream>>>(
      cond, rnn, emb, b0, b1, b2, bih, bhh, Wc, bc, Wa, ba, lines, actions,
      W0h, W1h, W2h, Wihh, Whhh, out);
}
